// GNNEncoder_15496242004301
// MI455X (gfx1250) — matrix operand order repaired and checked by reading, not compiled, not run on hardware
//
#include <hip/hip_runtime.h>
#include <hip/hip_bf16.h>
#include <cstddef>

// ---------------------------------------------------------------------------
// GNNEncoder for MI455X (gfx1250).
//
// Math collapse (exact, not an approximation):
//  * fam cross-attn keys/values are broadcast(gf) -> uniform softmax ->
//    feat[b,n,:] == gf2[b] @ fam_vw^T + fam_vb  (constant over n; q/k dead)
//  * arm cross-attn inputs are constant rows     -> uniform softmax ->
//    fe[b,e,:]  == feat[b] @ arm_vw^T + arm_vb   (constant over e)
//  * edge0[b,c] = relu(BN_b(fe @ ep_w^T + ep_b)) (stats reduce to 4 batches)
// Real work = 2 message-passing steps:
//    a=xv@A^T, bb=xv@B^T, u=xv@U^T, vv=xv@V^T   (256x256x256 GEMMs)
//    eE = edge@E^T                              (16384x256x256 GEMM, 2.1 GFLOP)
//    edge += relu(BN(a_i + bb_j + eE));  e=softmax_j(sigmoid(edge));
//    agg = sum_j e*vv_j / N;  xv = relu(xv + BN(u+agg))
// All GEMMs via v_wmma_f32_16x16x32_f16 (f16 in, f32 accumulate).
// All reductions sequential/2-stage -> deterministic across graph replays.
// ---------------------------------------------------------------------------

#define B_   4
#define N_   64
#define D_   256
#define NN_  (N_ * N_)      // 4096
#define BNN_ (B_ * NN_)     // 16384
#define EPS_ 1e-5f

typedef __attribute__((ext_vector_type(16))) _Float16 v16h;
typedef __attribute__((ext_vector_type(8)))  _Float16 v8h;
typedef __attribute__((ext_vector_type(8)))  float    v8f;

// -------------------------- WMMA GEMM: C = X * W^T --------------------------
// X: [M,K] row-major f16.  W: [Nc,K] row-major f16.  C: [M,Nc] f32.
// A fragment (16x32): lane L -> row m0+(L&15); halfs h -> K=(L>>4)*16+h
__device__ __forceinline__ v16h load_fragA(const _Float16* __restrict__ X,
                                           int m0, int k0, int ld, int lane) {
  const int r  = m0 + (lane & 15);
  const int kb = k0 + ((lane >> 4) << 4);
  const _Float16* p = X + (size_t)r * ld + kb;
  v8h lo = *(const v8h*)(p);        // K = kb .. kb+7
  v8h hi = *(const v8h*)(p + 8);    // K = kb+8 .. kb+15
  return __builtin_shufflevector(lo, hi, 0,1,2,3,4,5,6,7,8,9,10,11,12,13,14,15);
}
// B fragment (32x16 of W^T) == rows of W: lane L -> col n0+(L&15);
// halfs h -> K = k0 + (L>>4)*16 + h  (contiguous 32B per lane)
__device__ __forceinline__ v16h load_fragB(const _Float16* __restrict__ W,
                                           int n0, int k0, int ld, int lane) {
  const int n  = n0 + (lane & 15);
  const int kb = k0 + ((lane >> 4) << 4);
  return *(const v16h*)(W + (size_t)n * ld + kb);
}

__global__ __launch_bounds__(256) void wmma_gemm_xwt(
    const _Float16* __restrict__ X, const _Float16* __restrict__ W,
    float* __restrict__ C, int M, int Nc, int K) {
  const int lane = threadIdx.x & 31;
  const int wave = threadIdx.x >> 5;
  const int n0   = blockIdx.x * 16;
  const int m0   = (blockIdx.y * 8 + wave) * 16;   // uniform per wave -> EXEC stays full
  if (m0 >= M) return;
  v8f acc = {};
#pragma unroll
  for (int k0 = 0; k0 < 256; k0 += 32) {           // K == 256 always here
    v16h af = load_fragA(X, m0, k0, K, lane);
    v16h bf = load_fragB(W, n0, k0, K, lane);
    acc = __builtin_amdgcn_wmma_f32_16x16x32_f16(false, af, false, bf,
                                                 (short)0, acc, false, false);
  }
  const int col   = n0 + (lane & 15);
  const int rbase = m0 + ((lane >> 4) << 3);
#pragma unroll
  for (int r = 0; r < 8; ++r)
    C[(size_t)(rbase + r) * Nc + col] = acc[r];
}

// ------------------------------ f32 -> f16 ---------------------------------
__global__ __launch_bounds__(256) void f32_to_f16_kernel(
    const float* __restrict__ in, _Float16* __restrict__ out, int n) {
  int i = blockIdx.x * 256 + threadIdx.x;
  if (i < n) out[i] = (_Float16)in[i];
}

// ------------------------------ front end ----------------------------------
// One block, 256 threads (thread = channel d). Produces edge0[b,c].
__global__ __launch_bounds__(256) void front_kernel(
    const float* __restrict__ x,
    const float* __restrict__ w1, const float* __restrict__ b1,
    const float* __restrict__ gn1g, const float* __restrict__ gn1b,
    const float* __restrict__ w2,  const float* __restrict__ b2,
    const float* __restrict__ gn2g, const float* __restrict__ gn2b,
    const float* __restrict__ fam_vw, const float* __restrict__ fam_vb,
    const float* __restrict__ arm_vw, const float* __restrict__ arm_vb,
    const float* __restrict__ ep_w, const float* __restrict__ ep_b,
    const float* __restrict__ ep_g, const float* __restrict__ ep_beta,
    float* __restrict__ edge0) {
  __shared__ float sA[B_][D_];
  __shared__ float sB[B_][D_];
  const int d = threadIdx.x;
  float g[B_];
  // depthwise-over-N conv, kernel 3, SAME (cross-correlation like lax.conv)
#pragma unroll
  for (int b = 0; b < B_; ++b) {
    float acc = b1[0];
    const float* xb = x + (size_t)b * N_ * D_;
    for (int n = 0; n < N_; ++n) {
      const float* xr = xb + n * D_;
      float xm = (d > 0)       ? xr[d - 1] : 0.f;
      float xc = xr[d];
      float xp = (d < D_ - 1)  ? xr[d + 1] : 0.f;
      acc += xm * w1[n * 3 + 0] + xc * w1[n * 3 + 1] + xp * w1[n * 3 + 2];
    }
    g[b] = acc;
  }
  {  // BN over batch (4 samples) + relu + scalar affine
    float m = 0.25f * (g[0] + g[1] + g[2] + g[3]);
    float v = 0.f;
#pragma unroll
    for (int b = 0; b < B_; ++b) { float t = g[b] - m; v += t * t; }
    float rs = rsqrtf(v * 0.25f + EPS_);
#pragma unroll
    for (int b = 0; b < B_; ++b)
      g[b] = fmaxf(gn1g[d] * (g[b] - m) * rs + gn1b[d], 0.f) * w2[0] + b2[0];
  }
  {  // BN2 + relu -> sA
    float m = 0.25f * (g[0] + g[1] + g[2] + g[3]);
    float v = 0.f;
#pragma unroll
    for (int b = 0; b < B_; ++b) { float t = g[b] - m; v += t * t; }
    float rs = rsqrtf(v * 0.25f + EPS_);
#pragma unroll
    for (int b = 0; b < B_; ++b)
      sA[b][d] = fmaxf(gn2g[d] * (g[b] - m) * rs + gn2b[d], 0.f);
  }
  __syncthreads();
  // feat = sA @ fam_vw^T + fam_vb  -> sB
#pragma unroll
  for (int b = 0; b < B_; ++b) {
    float acc = fam_vb[d];
    const float* wr = fam_vw + (size_t)d * D_;
    for (int k = 0; k < D_; ++k) acc += sA[b][k] * wr[k];
    sB[b][d] = acc;
  }
  __syncthreads();
  // fe = sB @ arm_vw^T + arm_vb -> sA
#pragma unroll
  for (int b = 0; b < B_; ++b) {
    float acc = arm_vb[d];
    const float* wr = arm_vw + (size_t)d * D_;
    for (int k = 0; k < D_; ++k) acc += sB[b][k] * wr[k];
    sA[b][d] = acc;
  }
  __syncthreads();
  // fev = sA @ ep_w^T + ep_b ; BN over b ; relu -> edge0
#pragma unroll
  for (int b = 0; b < B_; ++b) {
    float acc = ep_b[d];
    const float* wr = ep_w + (size_t)d * D_;
    for (int k = 0; k < D_; ++k) acc += sA[b][k] * wr[k];
    g[b] = acc;
  }
  {
    float m = 0.25f * (g[0] + g[1] + g[2] + g[3]);
    float v = 0.f;
#pragma unroll
    for (int b = 0; b < B_; ++b) { float t = g[b] - m; v += t * t; }
    float rs = rsqrtf(v * 0.25f + EPS_);
#pragma unroll
    for (int b = 0; b < B_; ++b)
      edge0[b * D_ + d] = fmaxf(ep_g[d] * (g[b] - m) * rs + ep_beta[d], 0.f);
  }
}

// --------------------- broadcast edge0 -> full edge tensor -----------------
__global__ __launch_bounds__(256) void broadcast_edge_kernel(
    const float* __restrict__ e0, float* __restrict__ edge,
    _Float16* __restrict__ edge_h) {
  size_t gid = (size_t)blockIdx.x * 256 + threadIdx.x;  // < 2^22
  int c = (int)(gid & (D_ - 1));
  int b = (int)(gid >> 20);                             // NN_*D_ = 2^20
  float v = e0[b * D_ + c];
  edge[gid] = v;
  edge_h[gid] = (_Float16)v;
}

// --------- temp = a_i + bb_j + eE : two-stage BN stats (deterministic) ------
__global__ __launch_bounds__(256) void temp_stats_partial(
    const float* __restrict__ a, const float* __restrict__ bb,
    const float* __restrict__ eE, float* __restrict__ partial) {
  const int c  = threadIdx.x;
  const int r0 = blockIdx.x * 128;           // 128 blocks x 128 rows = 16384
  float s = 0.f, s2 = 0.f;
  for (int rr = 0; rr < 128; ++rr) {
    int r = r0 + rr;
    int b = r >> 12;                          // NN_ = 4096
    int e = r & (NN_ - 1);
    int i = e >> 6, j = e & (N_ - 1);
    float v = a[((b << 6) + i) * D_ + c] + bb[((b << 6) + j) * D_ + c]
            + eE[(size_t)r * D_ + c];
    s += v; s2 += v * v;
  }
  partial[blockIdx.x * 512 + c]       = s;
  partial[blockIdx.x * 512 + 256 + c] = s2;
}

__global__ __launch_bounds__(256) void stats_finalize(
    const float* __restrict__ partial, int nblk,
    float* __restrict__ mean, float* __restrict__ rstd, float invCount) {
  const int c = threadIdx.x;
  float s = 0.f, s2 = 0.f;
  for (int k = 0; k < nblk; ++k) {
    s  += partial[k * 512 + c];
    s2 += partial[k * 512 + 256 + c];
  }
  float m = s * invCount;
  float v = s2 * invCount - m * m;
  mean[c] = m;
  rstd[c] = rsqrtf(v + EPS_);
}

// ----------------- edge += relu(BN(temp)); re-encode f16 -------------------
__global__ __launch_bounds__(256) void edge_update_kernel(
    const float* __restrict__ a, const float* __restrict__ bb,
    const float* __restrict__ eE, const float* __restrict__ mean,
    const float* __restrict__ rstd, const float* __restrict__ bng,
    const float* __restrict__ bnb, float* __restrict__ edge,
    _Float16* __restrict__ edge_h) {
  const int c = threadIdx.x;
  const int r = blockIdx.x;                   // 16384 rows
  int b = r >> 12;
  int e = r & (NN_ - 1);
  int i = e >> 6, j = e & (N_ - 1);
  float v = a[((b << 6) + i) * D_ + c] + bb[((b << 6) + j) * D_ + c]
          + eE[(size_t)r * D_ + c];
  float t  = bng[c] * (v - mean[c]) * rstd[c] + bnb[c];
  size_t idx = (size_t)r * D_ + c;
  float ev = edge[idx] + fmaxf(t, 0.f);
  edge[idx]   = ev;
  edge_h[idx] = (_Float16)ev;
}

// --------- agg[b,i,c] = sum_j softmax_j(sigmoid(edge)) * vv[b,j,c] / N ------
__global__ __launch_bounds__(256) void softmax_agg_kernel(
    const float* __restrict__ edge, const float* __restrict__ vv,
    float* __restrict__ agg) {
  const int c  = threadIdx.x;
  const int bi = blockIdx.x;                  // 256 = B_*N_
  const int b  = bi >> 6, i = bi & (N_ - 1);
  const float* erow = edge + ((size_t)(b * NN_ + i * N_)) * D_ + c;
  const float* vrow = vv + ((size_t)(b * N_)) * D_ + c;
  float num = 0.f, den = 0.f;
  for (int j = 0; j < N_; ++j) {
    float sg = 1.f / (1.f + expf(-erow[(size_t)j * D_]));
    float es = expf(sg);                      // sigmoid in (0,1): stable
    den += es;
    num += es * vrow[(size_t)j * D_];
  }
  agg[(size_t)bi * D_ + c] = num / (den * (float)N_);
}

// ----------- xv_out = relu(res + BN(u+agg)) [+ x]; f16 side copy -----------
__global__ __launch_bounds__(256) void xv_update_kernel(
    const float* __restrict__ u, const float* __restrict__ agg,
    const float* __restrict__ res, const float* __restrict__ bng,
    const float* __restrict__ bnb, const float* __restrict__ addx,
    float* __restrict__ out, _Float16* __restrict__ outh) {
  const int c = threadIdx.x;
  float s = 0.f, s2 = 0.f;
  for (int r = 0; r < B_ * N_; ++r) {
    float t = u[r * D_ + c] + agg[r * D_ + c];
    s += t; s2 += t * t;
  }
  const float inv = 1.f / (float)(B_ * N_);
  float m  = s * inv;
  float va = s2 * inv - m * m;
  float rs = rsqrtf(va + EPS_);
  for (int r = 0; r < B_ * N_; ++r) {
    float t   = u[r * D_ + c] + agg[r * D_ + c];
    float bnv = bng[c] * (t - m) * rs + bnb[c];
    float val = fmaxf(res[r * D_ + c] + bnv, 0.f);
    out[r * D_ + c] = val + (addx ? addx[r * D_ + c] : 0.f);
    if (outh) outh[r * D_ + c] = (_Float16)val;
  }
}

// --------------------------------- host ------------------------------------
enum {
  IN_X = 0, IN_FAM_QW, IN_FAM_QB, IN_FAM_KW, IN_FAM_KB, IN_FAM_VW, IN_FAM_VB,
  IN_ARM_QW, IN_ARM_QB, IN_ARM_KW, IN_ARM_KB, IN_ARM_VW, IN_ARM_VB,
  IN_EP_W, IN_EP_B, IN_EP_G, IN_EP_BETA,
  IN_GFE1_W, IN_GFE1_B, IN_GFE2_W, IN_GFE2_B,
  IN_GN1_G, IN_GN1_B, IN_GN2_G, IN_GN2_B,
  IN_BNE1_G, IN_BNE1_B, IN_BNV1_G, IN_BNV1_B,
  IN_BNE2_G, IN_BNE2_B, IN_BNV2_G, IN_BNV2_B,
  IN_A1, IN_B1, IN_E1, IN_U1, IN_V1,
  IN_A2, IN_B2, IN_E2, IN_U2, IN_V2
};

extern "C" void kernel_launch(void* const* d_in, const int* in_sizes, int n_in,
                              void* d_out, int out_size, void* d_ws, size_t ws_size,
                              hipStream_t stream) {
  (void)in_sizes; (void)n_in; (void)out_size; (void)ws_size;
  const float* x = (const float*)d_in[IN_X];

  // ---- workspace carve (all regions fully rewritten every call) ----
  char* p = (char*)d_ws;
  auto carve = [&](size_t bytes) -> char* {
    char* r = p; p += (bytes + 255) & ~(size_t)255; return r;
  };
  float*    edge0   = (float*)carve(B_ * D_ * sizeof(float));
  float*    a       = (float*)carve(B_ * N_ * D_ * sizeof(float));
  float*    bbuf    = (float*)carve(B_ * N_ * D_ * sizeof(float));
  float*    u       = (float*)carve(B_ * N_ * D_ * sizeof(float));
  float*    vv      = (float*)carve(B_ * N_ * D_ * sizeof(float));
  float*    agg     = (float*)carve(B_ * N_ * D_ * sizeof(float));
  float*    xv1     = (float*)carve(B_ * N_ * D_ * sizeof(float));
  float*    eE      = (float*)carve((size_t)BNN_ * D_ * sizeof(float));
  float*    edge    = (float*)carve((size_t)BNN_ * D_ * sizeof(float));
  float*    partial = (float*)carve(128 * 512 * sizeof(float));
  float*    mean    = (float*)carve(D_ * sizeof(float));
  float*    rstd    = (float*)carve(D_ * sizeof(float));
  _Float16* xh      = (_Float16*)carve(B_ * N_ * D_ * sizeof(_Float16));
  _Float16* edge_h  = (_Float16*)carve((size_t)BNN_ * D_ * sizeof(_Float16));
  _Float16* wh[10];
  for (int i = 0; i < 10; ++i)
    wh[i] = (_Float16*)carve(D_ * D_ * sizeof(_Float16));

  // ---- f16 encodes: 10 weight matrices + x ----
  const int wIdx[10] = {IN_A1, IN_B1, IN_E1, IN_U1, IN_V1,
                        IN_A2, IN_B2, IN_E2, IN_U2, IN_V2};
  for (int i = 0; i < 10; ++i)
    f32_to_f16_kernel<<<(D_ * D_) / 256, 256, 0, stream>>>(
        (const float*)d_in[wIdx[i]], wh[i], D_ * D_);
  f32_to_f16_kernel<<<(B_ * N_ * D_) / 256, 256, 0, stream>>>(x, xh, B_ * N_ * D_);

  // ---- collapsed front-end -> edge0[b,c], broadcast to full edge tensor ----
  front_kernel<<<1, 256, 0, stream>>>(
      x,
      (const float*)d_in[IN_GFE1_W], (const float*)d_in[IN_GFE1_B],
      (const float*)d_in[IN_GN1_G],  (const float*)d_in[IN_GN1_B],
      (const float*)d_in[IN_GFE2_W], (const float*)d_in[IN_GFE2_B],
      (const float*)d_in[IN_GN2_G],  (const float*)d_in[IN_GN2_B],
      (const float*)d_in[IN_FAM_VW], (const float*)d_in[IN_FAM_VB],
      (const float*)d_in[IN_ARM_VW], (const float*)d_in[IN_ARM_VB],
      (const float*)d_in[IN_EP_W],   (const float*)d_in[IN_EP_B],
      (const float*)d_in[IN_EP_G],   (const float*)d_in[IN_EP_BETA],
      edge0);
  broadcast_edge_kernel<<<(BNN_ * D_) / 256, 256, 0, stream>>>(edge0, edge, edge_h);

  // ---- two message-passing steps ----
  const float* res = x;
  for (int s = 0; s < 2; ++s) {
    _Float16* Ah = wh[s * 5 + 0];
    _Float16* Bh = wh[s * 5 + 1];
    _Float16* Eh = wh[s * 5 + 2];
    _Float16* Uh = wh[s * 5 + 3];
    _Float16* Vh = wh[s * 5 + 4];
    const float* bneg = (const float*)d_in[s ? IN_BNE2_G : IN_BNE1_G];
    const float* bneb = (const float*)d_in[s ? IN_BNE2_B : IN_BNE1_B];
    const float* bnvg = (const float*)d_in[s ? IN_BNV2_G : IN_BNV1_G];
    const float* bnvb = (const float*)d_in[s ? IN_BNV2_B : IN_BNV1_B];

    dim3 gs(D_ / 16, (B_ * N_) / 128);   // (16, 2): node-feature GEMMs
    wmma_gemm_xwt<<<gs, 256, 0, stream>>>(xh, Ah, a,    B_ * N_, D_, D_);
    wmma_gemm_xwt<<<gs, 256, 0, stream>>>(xh, Bh, bbuf, B_ * N_, D_, D_);
    wmma_gemm_xwt<<<gs, 256, 0, stream>>>(xh, Uh, u,    B_ * N_, D_, D_);
    wmma_gemm_xwt<<<gs, 256, 0, stream>>>(xh, Vh, vv,   B_ * N_, D_, D_);

    dim3 ge(D_ / 16, BNN_ / 128);        // (16, 128): edge GEMM, 2.1 GFLOP
    wmma_gemm_xwt<<<ge, 256, 0, stream>>>(edge_h, Eh, eE, BNN_, D_, D_);

    temp_stats_partial<<<128, 256, 0, stream>>>(a, bbuf, eE, partial);
    stats_finalize<<<1, 256, 0, stream>>>(partial, 128, mean, rstd, 1.f / (float)BNN_);
    edge_update_kernel<<<BNN_, 256, 0, stream>>>(a, bbuf, eE, mean, rstd,
                                                 bneg, bneb, edge, edge_h);
    softmax_agg_kernel<<<B_ * N_, 256, 0, stream>>>(edge, vv, agg);

    if (s == 0) {
      xv_update_kernel<<<1, 256, 0, stream>>>(u, agg, res, bnvg, bnvb,
                                              nullptr, xv1, xh);
      res = xv1;
    } else {
      xv_update_kernel<<<1, 256, 0, stream>>>(u, agg, res, bnvg, bnvb,
                                              x, (float*)d_out, nullptr);
    }
  }
}
